// init_e_55387898250013
// MI455X (gfx1250) — matrix operand order repaired and checked by reading, compiled, not run on hardware
//
#include <hip/hip_runtime.h>
#include <hip/hip_bf16.h>

typedef __bf16 bf16_t;
typedef __attribute__((ext_vector_type(16))) __bf16 v16bf;
typedef __attribute__((ext_vector_type(8)))  float  v8f;

constexpr int H = 128;
constexpr int SBF = 136;                 // padded LDS row stride (bf16 elems): 272B, 16B-aligned, rotates banks
constexpr long long NNODES = 100000;
constexpr long long NEDGES = 800000;
constexpr int VOCAB = 95;

__device__ __forceinline__ float swishf(float v) {
    return v * (1.0f / (1.0f + __expf(-v)));
}

__device__ __forceinline__ unsigned pack2bf(float a, float b) {
    union { __bf16 h[2]; unsigned u; } t;
    t.h[0] = (__bf16)a; t.h[1] = (__bf16)b;
    return t.u;
}

// Low 32 bits of a generic pointer to __shared__ = LDS byte address (aperture in high bits).
__device__ __forceinline__ unsigned lds_addr(const void* p) {
    return (unsigned)(uintptr_t)p;
}

// GLOBAL_LOAD_ASYNC_TO_LDS_B128, GVS mode: lds-addr VGPR, 32-bit voffset VGPR, 64-bit SGPR base.
// Tracked by ASYNCcnt; bypasses VGPRs entirely.
__device__ __forceinline__ void async_b128(unsigned lds, const void* gbase, unsigned voff) {
    asm volatile("global_load_async_to_lds_b128 %0, %1, %2"
                 :: "v"(lds), "v"(voff), "s"(gbase)
                 : "memory");
}
__device__ __forceinline__ void wait_async0() {
    asm volatile("s_wait_asynccnt 0x0" ::: "memory");
}

// A fragment: 16x32 bf16. lane: row = rowBase + (lane&15); hf = lane>>4.
// elements 0..7  -> K = kb*32 + hf*8 .. +7
// elements 8..15 -> K = kb*32 + 16 + hf*8 .. +7      (ISA 7.12.2, 16-bit A 16x32)
__device__ __forceinline__ v16bf load_a_frag(const bf16_t* m, int rowBase, int kb, int ln, int hf) {
    const bf16_t* p = m + (rowBase + ln) * SBF + kb * 32 + hf * 8;
    v16bf a;
    reinterpret_cast<uint4*>(&a)[0] = *reinterpret_cast<const uint4*>(p);
    reinterpret_cast<uint4*>(&a)[1] = *reinterpret_cast<const uint4*>(p + 16);
    return a;
}

// B fragment: 32x16 bf16 from LDS weight stored transposed [n][k].
// lane: col n = nt*16 + (lane&15); elements 0..7 -> K = kb*32 + hf*8 .. +7, elements 8..15 -> K = kb*32 + 16 + hf*8 .. +7
__device__ __forceinline__ v16bf load_b_frag(const bf16_t* m, int nt, int kb, int ln, int hf) {
    const bf16_t* p = m + (nt * 16 + ln) * SBF + kb * 32 + hf * 8;
    v16bf b;
    reinterpret_cast<uint4*>(&b)[0] = *reinterpret_cast<const uint4*>(p);
    reinterpret_cast<uint4*>(&b)[1] = *reinterpret_cast<const uint4*>(p + 16);
    return b;
}

// ---- prep: transpose W2/W4/W5 to bf16 [n][k] in workspace (one-time, 196KB) ----
__global__ void prep_weights_kernel(const float* __restrict__ W,
                                    bf16_t* __restrict__ W2t,
                                    bf16_t* __restrict__ W4t,
                                    bf16_t* __restrict__ W5t) {
    int idx = blockIdx.x * 256 + threadIdx.x;     // 3 * 16384 threads
    int which = idx >> 14;
    int e = idx & 16383;
    int n = e >> 7, k = e & 127;
    int base = (which == 0) ? H : (which == 1) ? 3 * H : 4 * H;
    float v = W[(long long)(base + k) * H + n];
    bf16_t* dst = (which == 0) ? W2t : (which == 1) ? W4t : W5t;
    dst[n * H + k] = (bf16_t)v;
}

// ---- prep: E1 = emb@W1, E3 = emb@W3 (95x128 each, tiny); also write tuple scalar ----
__global__ void prep_emb_kernel(const float* __restrict__ emb, const float* __restrict__ W,
                                float* __restrict__ E1, float* __restrict__ E3,
                                float* __restrict__ out) {
    int idx = blockIdx.x * 256 + threadIdx.x;     // 2 * 95 * 128 = 24320 threads exactly
    if (idx == 0) out[NEDGES * H] = 1.0f;         // second tuple element of reference
    int half = idx >= VOCAB * H;
    int e = idx - half * VOCAB * H;
    int v = e >> 7, n = e & 127;
    int base = half ? 2 * H : 0;                  // W1 rows [0,H), W3 rows [2H,3H)
    float acc = 0.f;
    for (int k = 0; k < H; ++k)
        acc += emb[v * H + k] * W[(long long)(base + k) * H + n];
    (half ? E3 : E1)[v * H + n] = acc;
}

// ---- node kernel: P = chi@W2 + E1[x], Q = chi@W4 + E3[x] (two passes to cap acc pressure) ----
__global__ __launch_bounds__(256) void node_kernel(const int* __restrict__ x,
                                                   const float* __restrict__ chi,
                                                   const bf16_t* __restrict__ W2t,
                                                   const bf16_t* __restrict__ W4t,
                                                   const float* __restrict__ E1,
                                                   const float* __restrict__ E3,
                                                   float* __restrict__ P,
                                                   float* __restrict__ Q) {
    __shared__ bf16_t chiS[128 * SBF];
    __shared__ bf16_t w2S[128 * SBF];
    __shared__ bf16_t w4S[128 * SBF];
    __shared__ int    xS[128];

    int tid = threadIdx.x;
    long long nodeBase = (long long)blockIdx.x * 128;

    // async DMA both weight panels into padded LDS (2048 x 16B chunks each)
#pragma unroll
    for (int it = 0; it < 8; ++it) {
        int chunk = tid + it * 256;
        int r = chunk >> 4, c0 = (chunk & 15) * 8;
        unsigned off = (unsigned)chunk * 16u;
        async_b128(lds_addr(&w2S[r * SBF + c0]), W2t, off);
        async_b128(lds_addr(&w4S[r * SBF + c0]), W4t, off);
    }
    if (tid < 128) {
        long long row = nodeBase + tid;
        xS[tid] = x[row < NNODES ? row : NNODES - 1];
    }
    // chi tile: float4 global load -> bf16 pack -> b64 LDS store (overlaps async weight DMA)
#pragma unroll
    for (int it = 0; it < 16; ++it) {
        int chunk = tid + it * 256;               // 4096 float4 chunks
        int r = chunk >> 5, c0 = (chunk & 31) * 4;
        long long row = nodeBase + r;
        if (row >= NNODES) row = NNODES - 1;
        float4 v = *reinterpret_cast<const float4*>(chi + row * H + c0);
        uint2 pk = { pack2bf(v.x, v.y), pack2bf(v.z, v.w) };
        *reinterpret_cast<uint2*>(&chiS[r * SBF + c0]) = pk;
    }
    wait_async0();
    __syncthreads();

    int wave = tid >> 5, lane = tid & 31, ln = lane & 15, hf = lane >> 4;
    int rowBase = wave * 16;

    // ---- pass 1: accP = chi@W2 ----
    {
        v8f acc[8] = {};
#pragma unroll
        for (int kb = 0; kb < 4; ++kb) {
            v16bf a = load_a_frag(chiS, rowBase, kb, ln, hf);
#pragma unroll
            for (int nt = 0; nt < 8; ++nt) {
                v16bf b2 = load_b_frag(w2S, nt, kb, ln, hf);
                acc[nt] = __builtin_amdgcn_wmma_f32_16x16x32_bf16(false, a, false, b2,
                                                                  (short)0, acc[nt], false, false);
            }
        }
#pragma unroll
        for (int r = 0; r < 8; ++r) {
            int mrow = rowBase + r + 8 * hf;      // C layout: M = r + 8*(lane>>4)
            long long node = nodeBase + mrow;
            if (node < NNODES) {
                const float* e1 = E1 + (long long)xS[mrow] * H;
#pragma unroll
                for (int nt = 0; nt < 8; ++nt) {
                    int c = nt * 16 + ln;
                    P[node * H + c] = acc[nt][r] + e1[c];
                }
            }
        }
    }
    // ---- pass 2: accQ = chi@W4 (stores of P overlap with this GEMM) ----
    {
        v8f acc[8] = {};
#pragma unroll
        for (int kb = 0; kb < 4; ++kb) {
            v16bf a = load_a_frag(chiS, rowBase, kb, ln, hf);
#pragma unroll
            for (int nt = 0; nt < 8; ++nt) {
                v16bf b4 = load_b_frag(w4S, nt, kb, ln, hf);
                acc[nt] = __builtin_amdgcn_wmma_f32_16x16x32_bf16(false, a, false, b4,
                                                                  (short)0, acc[nt], false, false);
            }
        }
#pragma unroll
        for (int r = 0; r < 8; ++r) {
            int mrow = rowBase + r + 8 * hf;
            long long node = nodeBase + mrow;
            if (node < NNODES) {
                const float* e3 = E3 + (long long)xS[mrow] * H;
#pragma unroll
                for (int nt = 0; nt < 8; ++nt) {
                    int c = nt * 16 + ln;
                    Q[node * H + c] = acc[nt][r] + e3[c];
                }
            }
        }
    }
}

// ---- edge kernel: e = swish( swish(rbf@Wrbf+brbf)@W5 + P[i] + Q[j] + b ) ----
__global__ __launch_bounds__(256) void edge_kernel(const float* __restrict__ rbf,
                                                   const int* __restrict__ iI,
                                                   const int* __restrict__ jI,
                                                   const float* __restrict__ Wrbf,
                                                   const float* __restrict__ brbf,
                                                   const float* __restrict__ bvec,
                                                   const bf16_t* __restrict__ W5t,
                                                   const float* __restrict__ P,
                                                   const float* __restrict__ Q,
                                                   float* __restrict__ out) {
    __shared__ bf16_t hS[128 * SBF];
    __shared__ bf16_t w5S[128 * SBF];
    __shared__ float  rbfS[128 * 6];
    __shared__ float  wrbfS[6 * 128];
    __shared__ float  brbfS[128];
    __shared__ float  bS[128];
    __shared__ int    iS[128], jS[128];

    int tid = threadIdx.x;
    long long edgeBase = (long long)blockIdx.x * 128;

    // async DMA W5 panel into padded LDS; drains only after h-compute below
#pragma unroll
    for (int it = 0; it < 8; ++it) {
        int chunk = tid + it * 256;               // 2048 x 16B
        int r = chunk >> 4, c0 = (chunk & 15) * 8;
        async_b128(lds_addr(&w5S[r * SBF + c0]), W5t, (unsigned)chunk * 16u);
    }
    for (int t = tid; t < 768; t += 256) {
        rbfS[t]  = rbf[edgeBase * 6 + t];
        wrbfS[t] = Wrbf[t];
    }
    if (tid < 128) {
        brbfS[tid] = brbf[tid];
        bS[tid]    = bvec[tid];
        int iE = iI[edgeBase + tid];
        int jE = jI[edgeBase + tid];
        iS[tid] = iE;
        jS[tid] = jE;
        // prefetch the gather rows (512B each) so they sit in L2/WGP$ by epilogue time
        const char* pr = (const char*)(P + (long long)iE * H);
        const char* qr = (const char*)(Q + (long long)jE * H);
#pragma unroll
        for (int o = 0; o < 512; o += 128) {
            __builtin_prefetch(pr + o, 0, 1);
            __builtin_prefetch(qr + o, 0, 1);
        }
    }
    __syncthreads();

    // h tile (128x128) from the 6-wide radial basis, bf16 into LDS (overlaps W5 async DMA)
#pragma unroll
    for (int it = 0; it < 32; ++it) {
        int p = tid + it * 256;
        int r = p >> 6, c0 = (p & 63) * 2;
        float a0 = brbfS[c0], a1 = brbfS[c0 + 1];
#pragma unroll
        for (int k = 0; k < 6; ++k) {
            float rv = rbfS[r * 6 + k];
            a0 += rv * wrbfS[k * 128 + c0];
            a1 += rv * wrbfS[k * 128 + c0 + 1];
        }
        *(unsigned*)(&hS[r * SBF + c0]) = pack2bf(swishf(a0), swishf(a1));
    }
    wait_async0();
    __syncthreads();

    int wave = tid >> 5, lane = tid & 31, ln = lane & 15, hf = lane >> 4;
    int rowBase = wave * 16;
    v8f acc[8] = {};
#pragma unroll
    for (int kb = 0; kb < 4; ++kb) {
        v16bf a = load_a_frag(hS, rowBase, kb, ln, hf);
#pragma unroll
        for (int nt = 0; nt < 8; ++nt) {
            v16bf b5 = load_b_frag(w5S, nt, kb, ln, hf);
            acc[nt] = __builtin_amdgcn_wmma_f32_16x16x32_bf16(false, a, false, b5,
                                                              (short)0, acc[nt], false, false);
        }
    }
#pragma unroll
    for (int r = 0; r < 8; ++r) {
        int mrow = rowBase + r + 8 * hf;
        long long e = edgeBase + mrow;
        const float* Pr = P + (long long)iS[mrow] * H;
        const float* Qr = Q + (long long)jS[mrow] * H;
        float* Or = out + e * H;
#pragma unroll
        for (int nt = 0; nt < 8; ++nt) {
            int c = nt * 16 + ln;
            Or[c] = swishf(acc[nt][r] + Pr[c] + Qr[c] + bS[c]);
        }
    }
}

extern "C" void kernel_launch(void* const* d_in, const int* in_sizes, int n_in,
                              void* d_out, int out_size, void* d_ws, size_t ws_size,
                              hipStream_t stream) {
    const int*   x    = (const int*)d_in[0];
    const float* chi  = (const float*)d_in[1];
    const float* rbf  = (const float*)d_in[2];
    const int*   iI   = (const int*)d_in[3];
    const int*   jI   = (const int*)d_in[4];
    const float* emb  = (const float*)d_in[5];
    const float* Wrbf = (const float*)d_in[6];
    const float* brbf = (const float*)d_in[7];
    const float* W    = (const float*)d_in[8];
    const float* b    = (const float*)d_in[9];
    float* out = (float*)d_out;

    // workspace layout: P (100000x128 f32) | Q | E1 (95x128) | E3 | W2t/W4t/W5t (128x128 bf16 each)
    float*  P   = (float*)d_ws;
    float*  Q   = P  + (size_t)NNODES * H;
    float*  E1  = Q  + (size_t)NNODES * H;
    float*  E3  = E1 + (size_t)VOCAB * H;
    bf16_t* W2t = (bf16_t*)(E3 + (size_t)VOCAB * H);
    bf16_t* W4t = W2t + H * H;
    bf16_t* W5t = W4t + H * H;

    prep_weights_kernel<<<192, 256, 0, stream>>>(W, W2t, W4t, W5t);          // 3*16384 elems
    prep_emb_kernel<<<95, 256, 0, stream>>>(emb, W, E1, E3, out);            // 2*95*128 elems
    int nodeBlocks = (int)((NNODES + 127) / 128);                            // 782
    node_kernel<<<nodeBlocks, 256, 0, stream>>>(x, chi, W2t, W4t, E1, E3, P, Q);
    int edgeBlocks = (int)(NEDGES / 128);                                    // 6250
    edge_kernel<<<edgeBlocks, 256, 0, stream>>>(rbf, iI, jI, Wrbf, brbf, b, W5t, P, Q, out);
}
